// CRFLSTMTagger_36223754174726
// MI455X (gfx1250) — matrix operand order repaired and checked by reading, not compiled, not run on hardware
//
#include <hip/hip_runtime.h>
#include <hip/hip_bf16.h>

// ---------------------------------------------------------------------------
// BiLSTM-CRF loss for MI455X (gfx1250, wave32, WMMA).
// GEMMs use v_wmma_f32_16x16x32_bf16; recurrence is a persistent per-direction
// workgroup with h in LDS; CRF scan is one workgroup per batch row.
// ---------------------------------------------------------------------------

typedef __attribute__((ext_vector_type(16))) __bf16 v16bf;
typedef __attribute__((ext_vector_type(8)))  __bf16 v8bf;
typedef __attribute__((ext_vector_type(8)))  float  v8f;

#define BATCH 64
#define TLEN  1024
#define DIN   512        // HID
#define HDIR_ 256
#define NGATE 1024       // 4*HDIR
#define NTAGS 128
#define NEGV  (-10000.0f)
#define BOS_  2

__device__ __forceinline__ __bf16 f2bf(float f) {
    unsigned u = __float_as_uint(f);
    u += 0x7fffu + ((u >> 16) & 1u);           // round-to-nearest-even
    unsigned short h = (unsigned short)(u >> 16);
    __bf16 r; __builtin_memcpy(&r, &h, 2); return r;
}
__device__ __forceinline__ float bf2f(__bf16 x) {
    unsigned short h; __builtin_memcpy(&h, &x, 2);
    return __uint_as_float(((unsigned)h) << 16);
}
__device__ __forceinline__ float sigmoidf_(float x) {
    return 1.0f / (1.0f + __expf(-x));
}
__device__ __forceinline__ float tanhf_(float x) {
    float e = __expf(-2.0f * fabsf(x));
    float r = (1.0f - e) / (1.0f + e);
    return copysignf(r, x);
}

// ---------------------------------------------------------------------------
// f32 -> bf16 conversion (weights)
// ---------------------------------------------------------------------------
__global__ void k_cvt_bf16(const float* __restrict__ src, __bf16* __restrict__ dst, int n) {
    int i = blockIdx.x * 256 + threadIdx.x;
    if (i < n) dst[i] = f2bf(src[i]);
}

// ---------------------------------------------------------------------------
// Embedding concat -> x0 (B*T, 512) bf16
// ---------------------------------------------------------------------------
__global__ void k_embed(const int* __restrict__ ss, const int* __restrict__ ws,
                        const int* __restrict__ ys,
                        const float* __restrict__ se, const float* __restrict__ we,
                        const float* __restrict__ ye, __bf16* __restrict__ X) {
    int row  = blockIdx.x * 8 + (threadIdx.x >> 5);
    int lane = threadIdx.x & 31;
    int sid = ss[row], wid = ws[row], yid = ys[row];
    __bf16* xr = X + (size_t)row * DIN;
    #pragma unroll
    for (int k = 0; k < 16; ++k) {
        int e = lane + 32 * k;
        float v;
        if (e < 128)      v = se[sid * 128 + e];
        else if (e < 256) v = we[wid * 128 + (e - 128)];
        else              v = ye[(size_t)yid * 256 + (e - 256)];
        xr[e] = f2bf(v);
    }
}

// ---------------------------------------------------------------------------
// GEMM: Y(M,N) = X(M,512) * W(N,512)^T + bias ; bf16 in, bf16 out.
// One wave computes a 16x64 tile: 4 f32 accumulators, K-loop of 16 WMMAs each.
// ---------------------------------------------------------------------------
__global__ void __launch_bounds__(256)
k_gemm_xp(const __bf16* __restrict__ X, const __bf16* __restrict__ W,
          const float* __restrict__ bias, __bf16* __restrict__ Y, int ntiles64) {
    int wid   = blockIdx.x * 8 + (threadIdx.x >> 5);
    int lane  = threadIdx.x & 31;
    int l15   = lane & 15;
    int lhi   = lane >> 4;
    int mtile = wid / ntiles64;
    int ntile = wid % ntiles64;
    int Nst   = ntiles64 * 64;

    const __bf16* xrow = X + (size_t)(mtile * 16 + l15) * DIN;

    v8f acc[4];
    #pragma unroll
    for (int s = 0; s < 4; ++s)
        #pragma unroll
        for (int v = 0; v < 8; ++v) acc[s][v] = 0.0f;

    #pragma unroll 4
    for (int kk = 0; kk < 16; ++kk) {
        int k0 = kk * 32 + lhi * 16;
        v8bf alo = *(const v8bf*)(xrow + k0);
        v8bf ahi = *(const v8bf*)(xrow + k0 + 8);
        v16bf a;
        #pragma unroll
        for (int i = 0; i < 8; ++i) { a[i] = alo[i]; a[8 + i] = ahi[i]; }
        int kb = kk * 32 + lhi * 16;
        #pragma unroll
        for (int s = 0; s < 4; ++s) {
            int n = ntile * 64 + s * 16 + l15;
            v16bf b = *(const v16bf*)(W + (size_t)n * DIN + kb);
            acc[s] = __builtin_amdgcn_wmma_f32_16x16x32_bf16(
                false, a, false, b, (short)0, acc[s], false, false);
        }
    }
    #pragma unroll
    for (int s = 0; s < 4; ++s) {
        int n = ntile * 64 + s * 16 + l15;
        float bv = bias[n];
        #pragma unroll
        for (int v = 0; v < 8; ++v) {
            int row = mtile * 16 + v + 8 * lhi;
            Y[(size_t)row * Nst + n] = f2bf(acc[s][v] + bv);
        }
    }
}

// Same GEMM but f32 output (logits), N = 128.
__global__ void __launch_bounds__(256)
k_gemm_logits(const __bf16* __restrict__ X, const __bf16* __restrict__ W,
              const float* __restrict__ bias, float* __restrict__ Y) {
    int wid   = blockIdx.x * 8 + (threadIdx.x >> 5);
    int lane  = threadIdx.x & 31;
    int l15   = lane & 15;
    int lhi   = lane >> 4;
    int mtile = wid >> 1;           // ntiles64 == 2
    int ntile = wid & 1;

    const __bf16* xrow = X + (size_t)(mtile * 16 + l15) * DIN;

    v8f acc[4];
    #pragma unroll
    for (int s = 0; s < 4; ++s)
        #pragma unroll
        for (int v = 0; v < 8; ++v) acc[s][v] = 0.0f;

    #pragma unroll 4
    for (int kk = 0; kk < 16; ++kk) {
        int k0 = kk * 32 + lhi * 16;
        v8bf alo = *(const v8bf*)(xrow + k0);
        v8bf ahi = *(const v8bf*)(xrow + k0 + 8);
        v16bf a;
        #pragma unroll
        for (int i = 0; i < 8; ++i) { a[i] = alo[i]; a[8 + i] = ahi[i]; }
        int kb = kk * 32 + lhi * 16;
        #pragma unroll
        for (int s = 0; s < 4; ++s) {
            int n = ntile * 64 + s * 16 + l15;
            v16bf b = *(const v16bf*)(W + (size_t)n * DIN + kb);
            acc[s] = __builtin_amdgcn_wmma_f32_16x16x32_bf16(
                false, a, false, b, (short)0, acc[s], false, false);
        }
    }
    #pragma unroll
    for (int s = 0; s < 4; ++s) {
        int n = ntile * 64 + s * 16 + l15;
        float bv = bias[n];
        #pragma unroll
        for (int v = 0; v < 8; ++v) {
            int row = mtile * 16 + v + 8 * lhi;
            Y[(size_t)row * NTAGS + n] = acc[s][v] + bv;
        }
    }
}

// ---------------------------------------------------------------------------
// LSTM recurrence. grid = 2 (direction), block = 1024 (32 waves).
// z = xp[:,t,:] + h @ Whh^T  (WMMA, h in LDS as bf16, padded rows)
// Each wave owns 2 (m-tile, j-tile) pairs; the 4 gate tiles of a pair share
// lane slots so i/f/g/o for one cell are register-local.
// ---------------------------------------------------------------------------
#define HPAD 264   // 256 + 8 bf16 -> 528-byte row stride, LDS-conflict-free

__global__ void __launch_bounds__(1024)
k_lstm(const __bf16* __restrict__ XP,   // (B*T, 2048) includes bias
       const __bf16* __restrict__ WHH,  // (L*2, 1024, 256)
       __bf16* __restrict__ XOUT,       // (B*T, 512)
       int layer) {
    __shared__ unsigned short hsh[BATCH * HPAD];

    const int dir  = blockIdx.x;
    const __bf16* Whh = WHH + (size_t)(layer * 2 + dir) * NGATE * HDIR_;
    const int tid = threadIdx.x, lane = tid & 31, wv = tid >> 5;
    const int l15 = lane & 15, lhi = lane >> 4;

    int mt[2], jt[2];
    #pragma unroll
    for (int p = 0; p < 2; ++p) { int tp = wv * 2 + p; mt[p] = tp >> 4; jt[p] = tp & 15; }

    for (int i = tid; i < BATCH * HPAD; i += 1024) hsh[i] = 0;

    float cst[2][8];
    #pragma unroll
    for (int p = 0; p < 2; ++p)
        #pragma unroll
        for (int v = 0; v < 8; ++v) cst[p][v] = 0.0f;

    __syncthreads();

    for (int t = 0; t < TLEN; ++t) {
        const int teff = dir ? (TLEN - 1 - t) : t;
        float hv[2][8];

        #pragma unroll
        for (int p = 0; p < 2; ++p) {
            const int bbase = mt[p] * 16;
            const int j     = jt[p] * 16 + l15;

            // init accumulators with xp (bias already folded in)
            v8f acc[4];
            #pragma unroll
            for (int g = 0; g < 4; ++g) {
                const int n = g * HDIR_ + j;
                #pragma unroll
                for (int v = 0; v < 8; ++v) {
                    int b = bbase + v + 8 * lhi;
                    acc[g][v] = bf2f(XP[((size_t)b * TLEN + teff) * 2048 + dir * NGATE + n]);
                }
            }
            // h @ Whh^T : K = 256
            const unsigned short* hrow = &hsh[(bbase + l15) * HPAD];
            #pragma unroll
            for (int kk = 0; kk < 8; ++kk) {
                int k0 = kk * 32 + lhi * 16;
                v8bf alo = *(const v8bf*)((const __bf16*)hrow + k0);
                v8bf ahi = *(const v8bf*)((const __bf16*)hrow + k0 + 8);
                v16bf a;
                #pragma unroll
                for (int i = 0; i < 8; ++i) { a[i] = alo[i]; a[8 + i] = ahi[i]; }
                int kb = kk * 32 + lhi * 16;
                #pragma unroll
                for (int g = 0; g < 4; ++g) {
                    int n = g * HDIR_ + j;
                    v16bf b = *(const v16bf*)(Whh + (size_t)n * HDIR_ + kb);
                    acc[g] = __builtin_amdgcn_wmma_f32_16x16x32_bf16(
                        false, a, false, b, (short)0, acc[g], false, false);
                }
            }
            // gate math (i, f, g, o) -> c, h
            #pragma unroll
            for (int v = 0; v < 8; ++v) {
                float ig = sigmoidf_(acc[0][v]);
                float fg = sigmoidf_(acc[1][v]);
                float gg = tanhf_(acc[2][v]);
                float og = sigmoidf_(acc[3][v]);
                float cc = fg * cst[p][v] + ig * gg;
                cst[p][v] = cc;
                hv[p][v] = og * tanhf_(cc);
            }
        }

        __syncthreads();   // all reads of h(t-1) done

        #pragma unroll
        for (int p = 0; p < 2; ++p) {
            const int bbase = mt[p] * 16;
            const int j     = jt[p] * 16 + l15;
            #pragma unroll
            for (int v = 0; v < 8; ++v) {
                int b = bbase + v + 8 * lhi;
                __bf16 hb = f2bf(hv[p][v]);
                ((__bf16*)hsh)[b * HPAD + j] = hb;
                XOUT[((size_t)b * TLEN + teff) * DIN + dir * HDIR_ + j] = hb;
            }
        }
        __syncthreads();   // h(t) visible for next step
    }
}

// ---------------------------------------------------------------------------
// Per-row log_softmax over 128 tags + length mask (in place). 1 wave per row.
// ---------------------------------------------------------------------------
__global__ void __launch_bounds__(256)
k_logsoftmax(float* __restrict__ TAG, const int* __restrict__ lens) {
    int row  = blockIdx.x * 8 + (threadIdx.x >> 5);
    int lane = threadIdx.x & 31;
    int b = row >> 10, t = row & (TLEN - 1);
    float* p = TAG + (size_t)row * NTAGS;
    float v[4];
    #pragma unroll
    for (int k = 0; k < 4; ++k) v[k] = p[lane + 32 * k];
    float m = fmaxf(fmaxf(v[0], v[1]), fmaxf(v[2], v[3]));
    #pragma unroll
    for (int d = 16; d; d >>= 1) m = fmaxf(m, __shfl_xor(m, d, 32));
    float s = 0.0f;
    #pragma unroll
    for (int k = 0; k < 4; ++k) s += __expf(v[k] - m);
    #pragma unroll
    for (int d = 16; d; d >>= 1) s += __shfl_xor(s, d, 32);
    float lse = m + __logf(s);
    float msk = (t < lens[b]) ? 1.0f : 0.0f;
    #pragma unroll
    for (int k = 0; k < 4; ++k) p[lane + 32 * k] = (v[k] - lse) * msk;
}

// ---------------------------------------------------------------------------
// CRF forward scan: one workgroup (128 threads) per batch row.
// trans transposed into LDS for conflict-free inner loop; early exit at len.
// ---------------------------------------------------------------------------
__global__ void __launch_bounds__(128)
k_crf(const float* __restrict__ TAG, const int* __restrict__ lens,
      const float* __restrict__ TRANS, float* __restrict__ Zpart) {
    __shared__ float trT[NTAGS * NTAGS];   // [j][i] = trans[i][j]
    __shared__ float zbuf[2][NTAGS];
    __shared__ float red[NTAGS];

    const int b = blockIdx.x, i = threadIdx.x;
    for (int e = i; e < NTAGS * NTAGS; e += 128)
        trT[(e & 127) * NTAGS + (e >> 7)] = TRANS[e];
    zbuf[0][i] = (i == BOS_) ? 0.0f : NEGV;
    __syncthreads();

    const int steps = min(TLEN - 1, lens[b]);
    int cur = 0;
    for (int s = 0; s < steps; ++s) {
        const float* Z = zbuf[cur];
        const float* tc = &trT[i];
        float m = -3.4e38f;
        for (int j = 0; j < NTAGS; ++j)
            m = fmaxf(m, Z[j] + tc[j * NTAGS]);
        float ssum = 0.0f;
        for (int j = 0; j < NTAGS; ++j)
            ssum += __expf(Z[j] + tc[j * NTAGS] - m);
        float emit = TAG[((size_t)b * TLEN + (s + 1)) * NTAGS + i];
        zbuf[cur ^ 1][i] = emit + m + __logf(ssum);
        __syncthreads();
        cur ^= 1;
    }
    // Zpart[b] = logsumexp over final Z
    red[i] = zbuf[cur][i];
    __syncthreads();
    for (int off = 64; off; off >>= 1) {
        if (i < off) red[i] = fmaxf(red[i], red[i + off]);
        __syncthreads();
    }
    float mm = red[0];
    __syncthreads();
    red[i] = __expf(zbuf[cur][i] - mm);
    __syncthreads();
    for (int off = 64; off; off >>= 1) {
        if (i < off) red[i] += red[i + off];
        __syncthreads();
    }
    if (i == 0) Zpart[b] = mm + __logf(red[0]);
}

// ---------------------------------------------------------------------------
// Gold score + final mean loss. 1 block, 256 threads (4 per batch row).
// ---------------------------------------------------------------------------
__global__ void __launch_bounds__(256)
k_final(const float* __restrict__ TAG, const int* __restrict__ tgts,
        const int* __restrict__ lens, const float* __restrict__ TRANS,
        const float* __restrict__ Zpart, float* __restrict__ out) {
    __shared__ float sacc[256];
    __shared__ float sB[64];
    const int tid = threadIdx.x, b = tid >> 2, sub = tid & 3;
    const int len = lens[b];
    float acc = 0.0f;
    for (int t = sub; t < TLEN; t += 4) {
        int tg   = tgts[b * TLEN + t];
        int prev = (t == 0) ? BOS_ : tgts[b * TLEN + t - 1];
        float e  = TAG[((size_t)b * TLEN + t) * NTAGS + tg];
        float tr = (t < len) ? TRANS[tg * NTAGS + prev] : 0.0f;
        acc += e + tr;
    }
    sacc[tid] = acc;
    __syncthreads();
    if (sub == 0) {
        float sc = sacc[tid] + sacc[tid + 1] + sacc[tid + 2] + sacc[tid + 3];
        sB[b] = Zpart[b] - sc;
    }
    __syncthreads();
    for (int off = 32; off; off >>= 1) {
        if (tid < off) sB[tid] += sB[tid + off];
        __syncthreads();
    }
    if (tid == 0) out[0] = sB[0] * (1.0f / 64.0f);
}

// ---------------------------------------------------------------------------
extern "C" void kernel_launch(void* const* d_in, const int* in_sizes, int n_in,
                              void* d_out, int out_size, void* d_ws, size_t ws_size,
                              hipStream_t stream) {
    (void)in_sizes; (void)n_in; (void)out_size; (void)ws_size;

    const int*   ss    = (const int*)  d_in[0];
    const int*   wbs   = (const int*)  d_in[1];
    const int*   ys    = (const int*)  d_in[2];
    const int*   lens  = (const int*)  d_in[3];
    const int*   tgts  = (const int*)  d_in[4];
    const float* se    = (const float*)d_in[5];
    const float* we    = (const float*)d_in[6];
    const float* ye    = (const float*)d_in[7];
    const float* wih   = (const float*)d_in[8];   // (2,2,1024,512)
    const float* whh   = (const float*)d_in[9];   // (2,2,1024,256)
    const float* blstm = (const float*)d_in[10];  // (2,2,1024)
    const float* wout  = (const float*)d_in[11];  // (128,512)
    const float* bout  = (const float*)d_in[12];
    const float* trans = (const float*)d_in[13];  // (128,128)

    char*  ws  = (char*)d_ws;
    size_t off = 0;
    auto alloc = [&](size_t bytes) -> void* {
        void* p = ws + off;
        off = (off + bytes + 255) & ~(size_t)255;
        return p;
    };
    const size_t NWIH = 2u * 2048u * 512u;   // 2,097,152
    const size_t NWHH = 4u * 1024u * 256u;   // 1,048,576
    const size_t NWO  = 128u * 512u;
    const size_t ROWS = (size_t)BATCH * TLEN;

    __bf16* wih_bf  = (__bf16*)alloc(NWIH * 2);
    __bf16* whh_bf  = (__bf16*)alloc(NWHH * 2);
    __bf16* wout_bf = (__bf16*)alloc(NWO * 2);
    __bf16* xA      = (__bf16*)alloc(ROWS * DIN * 2);
    __bf16* xB      = (__bf16*)alloc(ROWS * DIN * 2);
    __bf16* xp      = (__bf16*)alloc(ROWS * 2048 * 2);
    float*  tag     = (float*) alloc(ROWS * NTAGS * 4);
    float*  zpart   = (float*) alloc(64 * 4);

    // weight conversion
    k_cvt_bf16<<<(int)((NWIH + 255) / 256), 256, 0, stream>>>(wih, wih_bf, (int)NWIH);
    k_cvt_bf16<<<(int)((NWHH + 255) / 256), 256, 0, stream>>>(whh, whh_bf, (int)NWHH);
    k_cvt_bf16<<<(int)((NWO  + 255) / 256), 256, 0, stream>>>(wout, wout_bf, (int)NWO);

    // embeddings -> xA
    k_embed<<<8192, 256, 0, stream>>>(ss, wbs, ys, se, we, ye, xA);

    // 2 BiLSTM layers
    for (int l = 0; l < 2; ++l) {
        const __bf16* xin  = (l == 0) ? xA : xB;
        __bf16*       xout = (l == 0) ? xB : xA;
        // xp = xin @ Wih(l)^T + b : M=65536, N=2048 (fwd|bwd stacked), K=512
        // waves = (65536/16) * (2048/64) = 131072 -> 16384 blocks of 8 waves
        k_gemm_xp<<<16384, 256, 0, stream>>>(
            xin, wih_bf + (size_t)l * 2048 * 512, blstm + (size_t)l * 2048, xp, 32);
        // recurrence, both directions in parallel
        k_lstm<<<2, 1024, 0, stream>>>(xp, whh_bf, xout, l);
    }

    // logits = x2 @ Wout^T + b : waves = 4096*2 = 8192 -> 1024 blocks
    k_gemm_logits<<<1024, 256, 0, stream>>>(xA, wout_bf, bout, tag);

    // masked log_softmax (in place)
    k_logsoftmax<<<8192, 256, 0, stream>>>(tag, lens);

    // CRF forward partition function per batch row
    k_crf<<<64, 128, 0, stream>>>(tag, lens, trans, zpart);

    // gold score + mean loss
    k_final<<<1, 256, 0, stream>>>(tag, tgts, lens, trans, zpart, (float*)d_out);
}
